// MultiHeadAttention_51101520888319
// MI455X (gfx1250) — matrix operand order repaired and checked by reading, not compiled, not run on hardware
//
#include <hip/hip_runtime.h>
#include <hip/hip_bf16.h>

// Problem constants (match reference)
#define BB  4
#define SS  2048
#define DD  1024
#define HH  16
#define DKK 64
#define DVV 64

typedef _Float16 half_t;
typedef __attribute__((ext_vector_type(16))) _Float16 v16h;
typedef __attribute__((ext_vector_type(8)))  _Float16 v8h;
typedef __attribute__((ext_vector_type(8)))  float    v8f;
typedef __attribute__((ext_vector_type(4)))  unsigned int u32x4;
typedef __attribute__((ext_vector_type(8)))  unsigned int u32x8;

#define LDS_STRIDE 40   // halves; 80B row: 16B-aligned, conflict-free fragments
#define SC_STRIDE  72   // halves; 144B row for 64-wide score tiles

__device__ __forceinline__ v16h make_frag(v8h lo, v8h hi) {
  v16h r;
  #pragma unroll
  for (int j = 0; j < 8; ++j) { r[j] = lo[j]; r[j + 8] = hi[j]; }
  return r;
}

__device__ __forceinline__ v8f wmma_f16(v16h a, v16h b, v8f c) {
  return __builtin_amdgcn_wmma_f32_16x16x32_f16(false, a, false, b, (short)0, c,
                                                false, false);
}

// LDS byte offset of a shared-memory pointer (generic addr low 32 bits == LDS offset)
__device__ __forceinline__ unsigned lds_off(const void* p) {
  return (unsigned)(unsigned long long)p;
}

// Async global->LDS copy, 16B per lane. Tracked by ASYNCcnt.
__device__ __forceinline__ void async_copy_b128(unsigned ldsaddr, const void* gptr) {
  asm volatile("global_load_async_to_lds_b128 %0, %1, off"
               :: "v"(ldsaddr), "v"(gptr) : "memory");
}
__device__ __forceinline__ void async_wait0() {
  asm volatile("s_wait_asynccnt 0x0" ::: "memory");
}

// ---------------------------------------------------------------------------
// 128x64-tile GEMM:  out[M,N] = A[M,K] * Bw[K,N] + bias (+ residual)
// 8 waves, each owning a 32x32 output (4 accumulators): 2 A-frags x 2 B-frags
// -> 4 WMMAs per 32-K step, halving LDS traffic per FLOP vs 16x32/wave.
// A f32 path: convert to f16 into LDS manually. A f16 path: Tensor Data Mover
// pulls the 128x32-half tile into LDS; D# padding (pad_interval=16 DW,
// pad_amount=4 DW) recreates the 40-half LDS row stride in hardware.
// ---------------------------------------------------------------------------
template <bool A_HALF, bool OUT_HALF, bool RES>
__global__ __launch_bounds__(256) void gemm128x64(
    const void* __restrict__ Av, const float* __restrict__ Bw,
    const float* __restrict__ bias, const float* __restrict__ res,
    void* __restrict__ outv, int M, int N, int K) {
  __shared__ half_t sA[128 * LDS_STRIDE];  // [m][k]
  __shared__ half_t sB[64 * LDS_STRIDE];   // transposed: [n][k]
  const int tid = threadIdx.x;
  const int bm = blockIdx.y * 128;
  const int bn = blockIdx.x * 64;
  const int wave = tid >> 5, lane = tid & 31;
  const int wm = (wave >> 1) * 32, wn = (wave & 1) * 32;
  const int l16 = lane & 15, lhi = lane >> 4;
  const int kr = tid >> 3, nb = (tid & 7) * 8;  // B loader: 32 k-rows x 64 n
  const float*  Af = (const float*)Av;
  const half_t* Ah = (const half_t*)Av;
  v8f acc00 = {}, acc01 = {}, acc10 = {}, acc11 = {};

  for (int k0 = 0; k0 < K; k0 += 32) {
    if constexpr (A_HALF) {
      // TDM: one wave issues the 2-D tile descriptor; EXEC is ignored.
      if (wave == 0) {
        const unsigned long long ga =
            (unsigned long long)(const void*)(Ah + (size_t)bm * K + k0);
        u32x4 g0;
        g0.x = 1u;                                   // count=1 valid descriptor
        g0.y = lds_off(&sA[0]);                      // lds_addr (bytes)
        g0.z = (unsigned)ga;                         // global_addr[31:0]
        g0.w = (unsigned)((ga >> 32) & 0x01FFFFFFu)  // global_addr[56:32]
               | 0x80000000u;                        // type=2 ("image")
        u32x8 g1;
        g1.s0 = (1u << 16)    // data_size = 2 bytes
              | (1u << 20)    // pad_enable
              | (3u << 22)    // pad_interval: 16 DWORDs (32 halves of data)
              | (3u << 25);   // pad_amount: 4 DWORDs (8 halves)
        g1.s1 = ((unsigned)K & 0xFFFFu) << 16;           // tensor_dim0 lo16
        g1.s2 = ((unsigned)K >> 16) | (((unsigned)M & 0xFFFFu) << 16);
        g1.s3 = ((unsigned)M >> 16) | (32u << 16);       // tile_dim0 = 32
        g1.s4 = 128u;                                    // tile_dim1 = 128
        g1.s5 = (unsigned)K;                             // tensor_dim0_stride
        g1.s6 = 0u;
        g1.s7 = 0u;
        u32x4 gz = {0u, 0u, 0u, 0u};
        asm volatile("tensor_load_to_lds %0, %1, %2, %3"
                     :: "s"(g0), "s"(g1), "s"(gz), "s"(gz) : "memory");
        __builtin_amdgcn_s_wait_tensorcnt(0);
      }
    } else {
      // 512 chunks of 8 floats (128 rows x 32 cols), 2 per thread.
      #pragma unroll
      for (int i = 0; i < 2; ++i) {
        const int c = tid + i * 256;
        const int r = c >> 2, col = (c & 3) * 8;
        const float* src = Af + (size_t)(bm + r) * K + k0 + col;
        __builtin_prefetch(src + 32, 0, 1);
        v8h t;
        #pragma unroll
        for (int j = 0; j < 8; ++j) t[j] = (half_t)src[j];
        *(v8h*)&sA[r * LDS_STRIDE + col] = t;
      }
    }
    {
      const float* src = Bw + (size_t)(k0 + kr) * N + bn + nb;
      __builtin_prefetch(src + (size_t)32 * N, 0, 1);
      #pragma unroll
      for (int j = 0; j < 8; ++j) sB[(nb + j) * LDS_STRIDE + kr] = (half_t)src[j];
    }
    __syncthreads();
    const int kba = lhi * 8, kbb = lhi * 8;
    const half_t* pa0 = &sA[(wm + l16) * LDS_STRIDE + kba];
    const half_t* pa1 = &sA[(wm + 16 + l16) * LDS_STRIDE + kba];
    v16h a0 = make_frag(*(const v8h*)pa0, *(const v8h*)(pa0 + 16));
    v16h a1 = make_frag(*(const v8h*)pa1, *(const v8h*)(pa1 + 16));
    const half_t* pb0 = &sB[(wn + l16) * LDS_STRIDE + kbb];
    const half_t* pb1 = &sB[(wn + 16 + l16) * LDS_STRIDE + kbb];
    v16h b0 = make_frag(*(const v8h*)pb0, *(const v8h*)(pb0 + 16));
    v16h b1 = make_frag(*(const v8h*)pb1, *(const v8h*)(pb1 + 16));
    acc00 = wmma_f16(a0, b0, acc00);
    acc01 = wmma_f16(a0, b1, acc01);
    acc10 = wmma_f16(a1, b0, acc10);
    acc11 = wmma_f16(a1, b1, acc11);
    __syncthreads();
  }
  #pragma unroll
  for (int ms = 0; ms < 2; ++ms) {
    #pragma unroll
    for (int ns = 0; ns < 2; ++ns) {
      const v8f acc = ms ? (ns ? acc11 : acc10) : (ns ? acc01 : acc00);
      const int nc = bn + wn + ns * 16 + l16;
      const float bval = bias[nc];
      #pragma unroll
      for (int j = 0; j < 8; ++j) {
        const int mr = bm + wm + ms * 16 + lhi * 8 + j;
        float vv = acc[j] + bval;
        if (RES) vv += res[(size_t)mr * N + nc];
        if (OUT_HALF) ((half_t*)outv)[(size_t)mr * N + nc] = (half_t)vv;
        else          ((float*)outv)[(size_t)mr * N + nc]  = vv;
      }
    }
  }
}

// ---------------------------------------------------------------------------
// Attention scores: S = Qh * Kh^T with mask -> -inf. Q/K tiles (64 rows x 64
// halves) staged to LDS via async global->LDS b128 copies (no transpose
// needed: kh rows ARE the K^T fragments). Fragments then come from LDS.
// ---------------------------------------------------------------------------
__global__ __launch_bounds__(256) void attn_scores(
    const half_t* __restrict__ qh, const half_t* __restrict__ kh,
    const int* __restrict__ maskm, float* __restrict__ attn) {
  __shared__ half_t sQ[64 * SC_STRIDE];
  __shared__ half_t sK[64 * SC_STRIDE];
  const int kt = blockIdx.x * 64, qt = blockIdx.y * 64;
  const int bh = blockIdx.z, b = bh >> 4, h = bh & 15;
  const int tid = threadIdx.x, wave = tid >> 5, lane = tid & 31;
  const int wm = (wave >> 1) * 16, wn = (wave & 1) * 32;
  const int l16 = lane & 15, lhi = lane >> 4;
  const half_t* qb = qh + (size_t)b * SS * DD + (size_t)h * DKK;
  const half_t* kb = kh + (size_t)b * SS * DD + (size_t)h * DKK;

  // 512 chunks of 16B per tile, 2 per thread per tile, lane-contiguous.
  #pragma unroll
  for (int i = 0; i < 2; ++i) {
    const int c = tid + i * 256;
    const int r = c >> 3, col = (c & 7) * 8;
    async_copy_b128(lds_off(&sQ[r * SC_STRIDE + col]),
                    qb + (size_t)(qt + r) * DD + col);
    async_copy_b128(lds_off(&sK[r * SC_STRIDE + col]),
                    kb + (size_t)(kt + r) * DD + col);
  }
  async_wait0();
  __syncthreads();

  v8f acc0 = {}, acc1 = {};
  #pragma unroll
  for (int ks = 0; ks < 2; ++ks) {  // d = 0..31, 32..63
    const half_t* pa = &sQ[(wm + l16) * SC_STRIDE + ks * 32 + lhi * 8];
    v16h a = make_frag(*(const v8h*)pa, *(const v8h*)(pa + 16));
    const int kbb = ks * 32 + lhi * 8;
    const half_t* pb0 = &sK[(wn + l16) * SC_STRIDE + kbb];
    v16h b0 = make_frag(*(const v8h*)pb0, *(const v8h*)(pb0 + 16));
    acc0 = wmma_f16(a, b0, acc0);
    const half_t* pb1 = &sK[(wn + 16 + l16) * SC_STRIDE + kbb];
    v16h b1 = make_frag(*(const v8h*)pb1, *(const v8h*)(pb1 + 16));
    acc1 = wmma_f16(a, b1, acc1);
  }

  float* orow = attn + (size_t)(h * BB + b) * SS * SS;
  const int* mrow = maskm + (size_t)b * SS * SS;
  int mv[16];
  #pragma unroll
  for (int ns = 0; ns < 2; ++ns) {
    const int kc = kt + wn + ns * 16 + l16;
    #pragma unroll
    for (int j = 0; j < 8; ++j)
      mv[ns * 8 + j] = mrow[(size_t)(qt + wm + lhi * 8 + j) * SS + kc];
  }
  #pragma unroll
  for (int ns = 0; ns < 2; ++ns) {
    const v8f acc = ns ? acc1 : acc0;
    const int kc = kt + wn + ns * 16 + l16;
    #pragma unroll
    for (int j = 0; j < 8; ++j) {
      const int qr = qt + wm + lhi * 8 + j;
      orow[(size_t)qr * SS + kc] = mv[ns * 8 + j] ? acc[j] : -__builtin_inff();
    }
  }
}

// ---------------------------------------------------------------------------
// Row softmax over 2048-wide rows of attn, in place.
// ---------------------------------------------------------------------------
__global__ __launch_bounds__(256) void softmax_rows(float* __restrict__ attn) {
  __shared__ float red[256];
  const size_t row = blockIdx.x;
  float* p = attn + row * SS;
  const int tid = threadIdx.x;
  float v[8];
  float mx = -__builtin_inff();
  #pragma unroll
  for (int i = 0; i < 8; ++i) { v[i] = p[tid + i * 256]; mx = fmaxf(mx, v[i]); }
  red[tid] = mx;
  __syncthreads();
  for (int s = 128; s > 0; s >>= 1) {
    if (tid < s) red[tid] = fmaxf(red[tid], red[tid + s]);
    __syncthreads();
  }
  const float m = red[0];
  __syncthreads();
  float sum = 0.f;
  #pragma unroll
  for (int i = 0; i < 8; ++i) { const float e = __expf(v[i] - m); v[i] = e; sum += e; }
  red[tid] = sum;
  __syncthreads();
  for (int s = 128; s > 0; s >>= 1) {
    if (tid < s) red[tid] += red[tid + s];
    __syncthreads();
  }
  const float tot = red[0];
  const bool dead = !(m > -__builtin_inff()) || !(tot > 0.f);
  const float inv = dead ? 0.f : 1.0f / tot;
  #pragma unroll
  for (int i = 0; i < 8; ++i) p[tid + i * 256] = dead ? 0.f : v[i] * inv;
}

// ---------------------------------------------------------------------------
// ctx[b, q, h*64 + n] = sum_k attn[h*B+b, q, k] * vh[b, k, h*64 + n]
// 128-row q tile, each wave 32x32 -> 4 WMMAs per 32-K step.
// ---------------------------------------------------------------------------
__global__ __launch_bounds__(256) void attn_ctx(
    const float* __restrict__ attn, const half_t* __restrict__ vh,
    half_t* __restrict__ ctx) {
  __shared__ half_t sA[128 * LDS_STRIDE];
  __shared__ half_t sB[64 * LDS_STRIDE];
  const int qt = blockIdx.y * 128;
  const int bh = blockIdx.z, b = bh >> 4, h = bh & 15;
  const int tid = threadIdx.x, wave = tid >> 5, lane = tid & 31;
  const int wm = (wave >> 1) * 32, wn = (wave & 1) * 32;
  const int l16 = lane & 15, lhi = lane >> 4;
  const int kr = tid >> 3, nb = (tid & 7) * 8;
  const float*  arow = attn + (size_t)(h * BB + b) * SS * SS;
  const half_t* vb   = vh + (size_t)b * SS * DD + (size_t)h * DVV;
  v8f acc00 = {}, acc01 = {}, acc10 = {}, acc11 = {};
  for (int k0 = 0; k0 < SS; k0 += 32) {
    #pragma unroll
    for (int i = 0; i < 2; ++i) {      // attn tile: 128 rows x 32 cols
      const int c = tid + i * 256;
      const int r = c >> 2, col = (c & 3) * 8;
      const float* src = arow + (size_t)(qt + r) * SS + k0 + col;
      __builtin_prefetch(src + 32, 0, 1);
      v8h t;
      #pragma unroll
      for (int j = 0; j < 8; ++j) t[j] = (half_t)src[j];
      *(v8h*)&sA[r * LDS_STRIDE + col] = t;
    }
    {
      const half_t* src = vb + (size_t)(k0 + kr) * DD + nb;
      #pragma unroll
      for (int j = 0; j < 8; ++j) sB[(nb + j) * LDS_STRIDE + kr] = src[j];
    }
    __syncthreads();
    const int kba = lhi * 8, kbb = lhi * 8;
    const half_t* pa0 = &sA[(wm + l16) * LDS_STRIDE + kba];
    const half_t* pa1 = &sA[(wm + 16 + l16) * LDS_STRIDE + kba];
    v16h a0 = make_frag(*(const v8h*)pa0, *(const v8h*)(pa0 + 16));
    v16h a1 = make_frag(*(const v8h*)pa1, *(const v8h*)(pa1 + 16));
    const half_t* pb0 = &sB[(wn + l16) * LDS_STRIDE + kbb];
    const half_t* pb1 = &sB[(wn + 16 + l16) * LDS_STRIDE + kbb];
    v16h b0 = make_frag(*(const v8h*)pb0, *(const v8h*)(pb0 + 16));
    v16h b1 = make_frag(*(const v8h*)pb1, *(const v8h*)(pb1 + 16));
    acc00 = wmma_f16(a0, b0, acc00);
    acc01 = wmma_f16(a0, b1, acc01);
    acc10 = wmma_f16(a1, b0, acc10);
    acc11 = wmma_f16(a1, b1, acc11);
    __syncthreads();
  }
  #pragma unroll
  for (int ms = 0; ms < 2; ++ms) {
    #pragma unroll
    for (int ns = 0; ns < 2; ++ns) {
      const v8f acc = ms ? (ns ? acc11 : acc10) : (ns ? acc01 : acc00);
      const int nc = wn + ns * 16 + l16;  // 0..63 within DV
      #pragma unroll
      for (int j = 0; j < 8; ++j) {
        const int qr = qt + wm + ms * 16 + lhi * 8 + j;
        ctx[(size_t)(b * SS + qr) * DD + h * DVV + nc] = (half_t)acc[j];
      }
    }
  }
}

// ---------------------------------------------------------------------------
// LayerNorm over rows of 1024, scaled by mask[b,s].
// ---------------------------------------------------------------------------
__global__ __launch_bounds__(256) void layernorm_kernel(
    const float* __restrict__ x, const float* __restrict__ gamma,
    const float* __restrict__ beta, const float* __restrict__ mask,
    float* __restrict__ y) {
  __shared__ float red[256];
  const size_t row = blockIdx.x;
  const float* px = x + row * DD;
  const int tid = threadIdx.x;
  float v[4];
  float s = 0.f;
  #pragma unroll
  for (int i = 0; i < 4; ++i) { v[i] = px[tid + i * 256]; s += v[i]; }
  red[tid] = s;
  __syncthreads();
  for (int st = 128; st > 0; st >>= 1) {
    if (tid < st) red[tid] += red[tid + st];
    __syncthreads();
  }
  const float mu = red[0] * (1.0f / DD);
  __syncthreads();
  float d = 0.f;
  #pragma unroll
  for (int i = 0; i < 4; ++i) { const float t = v[i] - mu; d += t * t; }
  red[tid] = d;
  __syncthreads();
  for (int st = 128; st > 0; st >>= 1) {
    if (tid < st) red[tid] += red[tid + st];
    __syncthreads();
  }
  const float rstd = rsqrtf(red[0] * (1.0f / DD) + 1e-5f);
  const float mk = mask[row];
  #pragma unroll
  for (int i = 0; i < 4; ++i) {
    const int c = tid + i * 256;
    y[row * DD + c] = (gamma[c] * (v[i] - mu) * rstd + beta[c]) * mk;
  }
}

// ---------------------------------------------------------------------------
extern "C" void kernel_launch(void* const* d_in, const int* in_sizes, int n_in,
                              void* d_out, int out_size, void* d_ws,
                              size_t ws_size, hipStream_t stream) {
  (void)in_sizes; (void)n_in; (void)out_size; (void)ws_size;
  const float* q     = (const float*)d_in[0];
  const float* k     = (const float*)d_in[1];
  const float* v     = (const float*)d_in[2];
  const float* mask  = (const float*)d_in[3];
  const int*   maskm = (const int*)d_in[4];
  const float* Wq = (const float*)d_in[5];  const float* bq = (const float*)d_in[6];
  const float* Wk = (const float*)d_in[7];  const float* bk = (const float*)d_in[8];
  const float* Wv = (const float*)d_in[9];  const float* bv = (const float*)d_in[10];
  const float* Wo = (const float*)d_in[11]; const float* bo = (const float*)d_in[12];
  const float* gamma = (const float*)d_in[13];
  const float* beta  = (const float*)d_in[14];

  float* y    = (float*)d_out;
  float* attn = (float*)d_out + (size_t)BB * SS * DD;  // (H*B, S, S)

  char* ws = (char*)d_ws;
  half_t* qh   = (half_t*)(ws);                        // 16 MB (B,S,H*DK) f16
  half_t* kh   = (half_t*)(ws + ((size_t)16 << 20));   // 16 MB
  half_t* vh   = (half_t*)(ws + ((size_t)32 << 20));   // 16 MB
  half_t* ctx  = (half_t*)(ws + ((size_t)48 << 20));   // 16 MB (B,S,H*DV) f16
  float*  xres = (float*)(ws + ((size_t)64 << 20));    // 32 MB x = proj + q

  const dim3 blk(256);
  const dim3 gproj(DD / 64, (BB * SS) / 128);          // (16, 64)

  gemm128x64<false, true, false><<<gproj, blk, 0, stream>>>(q, Wq, bq, nullptr, qh, BB * SS, DD, DD);
  gemm128x64<false, true, false><<<gproj, blk, 0, stream>>>(k, Wk, bk, nullptr, kh, BB * SS, DD, DD);
  gemm128x64<false, true, false><<<gproj, blk, 0, stream>>>(v, Wv, bv, nullptr, vh, BB * SS, DD, DD);

  const dim3 gsc(SS / 64, SS / 64, BB * HH);           // (32, 32, 64)
  attn_scores<<<gsc, blk, 0, stream>>>(qh, kh, maskm, attn);

  softmax_rows<<<dim3(BB * HH * SS), blk, 0, stream>>>(attn);  // 131072 rows

  const dim3 gctx(1, SS / 128, BB * HH);               // (1, 16, 64)
  attn_ctx<<<gctx, blk, 0, stream>>>(attn, vh, ctx);

  gemm128x64<true, false, true><<<gproj, blk, 0, stream>>>(ctx, Wo, bo, q, xres, BB * SS, DD, DD);

  layernorm_kernel<<<dim3(BB * SS), blk, 0, stream>>>(xres, gamma, beta, mask, y);
}
